// PointnetFPModule_24481313587583
// MI455X (gfx1250) — matrix operand order repaired and checked by reading, not compiled, not run on hardware
//
#include <hip/hip_runtime.h>
#include <hip/hip_bf16.h>

typedef __bf16 bf16;
typedef __attribute__((ext_vector_type(16))) __bf16 v16bf;
typedef __attribute__((ext_vector_type(8)))  __bf16 v8bf;
typedef __attribute__((ext_vector_type(8)))  float  v8f;
typedef __attribute__((ext_vector_type(4)))  float  v4f;

// Problem constants (from reference setup_inputs)
#define BB   8
#define NP   8192      // unknown points per batch
#define MP   2048      // known points per batch
#define C2   256       // known_feats channels (first in concat)
#define CIN  512
#define OUT  512
#define MTOT (BB * NP) // 65536 global GEMM rows (points)

union Frag { v16bf v; v8bf h[2]; };

// CDNA5 async copy: 16B per lane, global -> LDS, tracked by ASYNCcnt.
// lds_off: LDS byte address in a VGPR (low 32 bits of flat LDS address),
// gaddr : 64-bit global address in a VGPR pair.
__device__ __forceinline__ void async_copy_b128(unsigned lds_off, const void* gaddr) {
    asm volatile("global_load_async_to_lds_b128 %0, %1, off"
                 :: "v"(lds_off), "v"((unsigned long long)(uintptr_t)gaddr)
                 : "memory");
}
__device__ __forceinline__ void wait_async0() {
    asm volatile("s_wait_asynccnt 0x0" ::: "memory");
}

// ---------------------------------------------------------------------------
// Kernel 1: fold BN into scale/shift, convert W1/W2 to bf16 (row-major [o][c])
// ---------------------------------------------------------------------------
__global__ void prep_kernel(const float* __restrict__ W1f, const float* __restrict__ W2f,
                            const float* __restrict__ g1, const float* __restrict__ b1,
                            const float* __restrict__ m1, const float* __restrict__ v1,
                            const float* __restrict__ g2, const float* __restrict__ b2,
                            const float* __restrict__ m2, const float* __restrict__ v2,
                            bf16* __restrict__ W1b, bf16* __restrict__ W2b,
                            float* __restrict__ s1, float* __restrict__ sh1,
                            float* __restrict__ s2, float* __restrict__ sh2) {
    int g = blockIdx.x * 256 + threadIdx.x;
    const int WN = CIN * OUT;  // 262144
    if (g < WN) {
        W1b[g] = (bf16)W1f[g];
    } else if (g < 2 * WN) {
        W2b[g - WN] = (bf16)W2f[g - WN];
    } else if (g < 2 * WN + OUT) {
        int o = g - 2 * WN;
        float s = g1[o] * rsqrtf(v1[o] + 1e-5f);
        s1[o] = s; sh1[o] = b1[o] - m1[o] * s;
    } else if (g < 2 * WN + 2 * OUT) {
        int o = g - 2 * WN - OUT;
        float s = g2[o] * rsqrtf(v2[o] + 1e-5f);
        s2[o] = s; sh2[o] = b2[o] - m2[o] * s;
    }
}

// ---------------------------------------------------------------------------
// Kernel 2: brute-force 3-NN (squared L2) + inverse-distance weights.
// known points for the batch live in LDS (24 KB); one thread per query point.
// ---------------------------------------------------------------------------
__global__ __launch_bounds__(256) void three_nn_kernel(
        const float* __restrict__ unknown, const float* __restrict__ known,
        int* __restrict__ idx, float* __restrict__ wgt) {
    __shared__ float kx[MP], ky[MP], kz[MP];
    const int b = blockIdx.y;
    const float* kb = known + (size_t)b * MP * 3;
    for (int j = threadIdx.x; j < MP; j += 256) {
        kx[j] = kb[j * 3 + 0]; ky[j] = kb[j * 3 + 1]; kz[j] = kb[j * 3 + 2];
    }
    __syncthreads();

    const int p = blockIdx.x * 256 + threadIdx.x;
    const size_t pi = (size_t)b * NP + p;
    const float ux = unknown[pi * 3 + 0];
    const float uy = unknown[pi * 3 + 1];
    const float uz = unknown[pi * 3 + 2];

    float d0 = 3.4e38f, d1 = 3.4e38f, d2 = 3.4e38f;
    int   i0 = 0, i1 = 0, i2 = 0;
    #pragma unroll 4
    for (int j = 0; j < MP; ++j) {
        float dx = ux - kx[j], dy = uy - ky[j], dz = uz - kz[j];
        float d = dx * dx + dy * dy + dz * dz;
        if (d < d2) {
            if (d < d1) {
                d2 = d1; i2 = i1;
                if (d < d0) { d1 = d0; i1 = i0; d0 = d; i0 = j; }
                else        { d1 = d;  i1 = j; }
            } else { d2 = d; i2 = j; }
        }
    }
    float r0 = 1.0f / (d0 + 1e-8f);
    float r1 = 1.0f / (d1 + 1e-8f);
    float r2 = 1.0f / (d2 + 1e-8f);
    float rs = 1.0f / (r0 + r1 + r2);
    idx[pi * 3 + 0] = i0; idx[pi * 3 + 1] = i1; idx[pi * 3 + 2] = i2;
    wgt[pi * 3 + 0] = r0 * rs; wgt[pi * 3 + 1] = r1 * rs; wgt[pi * 3 + 2] = r2 * rs;
}

// ---------------------------------------------------------------------------
// Kernel 3: interpolate known_feats + concat unknow_feats, emit bf16 GEMM
// input in POINT-MAJOR layout XinT[b*NP + p][c] (c contiguous, c<256 interp,
// c>=256 pass-through). Thread = (point, 16-channel slab).
// ---------------------------------------------------------------------------
__global__ __launch_bounds__(256) void build_input_kernel(
        const float* __restrict__ known_feats, const float* __restrict__ unknow_feats,
        const int* __restrict__ idx, const float* __restrict__ wgt,
        bf16* __restrict__ XinT) {
    const int b  = blockIdx.y;
    const int p  = blockIdx.x * 16 + (threadIdx.x >> 4);
    const int c0 = (threadIdx.x & 15) * 16;
    const size_t pi = (size_t)b * NP + p;
    const int   i0 = idx[pi * 3 + 0], i1 = idx[pi * 3 + 1], i2 = idx[pi * 3 + 2];
    const float w0 = wgt[pi * 3 + 0], w1 = wgt[pi * 3 + 1], w2 = wgt[pi * 3 + 2];

    const float* kf = known_feats  + (size_t)b * C2 * MP;
    const float* uf = unknow_feats + (size_t)b * C2 * NP;
    bf16* xr = XinT + pi * CIN;

    #pragma unroll
    for (int ci = 0; ci < 16; ++ci) {
        int c = c0 + ci;
        const float* kfc = kf + (size_t)c * MP;
        float v = w0 * kfc[i0] + w1 * kfc[i1] + w2 * kfc[i2];
        xr[c] = (bf16)v;
    }
    #pragma unroll
    for (int ci = 0; ci < 16; ++ci) {
        int c = c0 + ci;
        xr[C2 + c] = (bf16)uf[(size_t)c * NP + p];
    }
}

// ---------------------------------------------------------------------------
// Kernel 4: fused GEMM + BN + ReLU using v_wmma_f32_16x16x32_bf16 with
// double-buffered LDS stages filled by GLOBAL_LOAD_ASYNC_TO_LDS_B128.
//   Yt[row][o] = relu(scale[o] * sum_k A[row][k] * W[o][k] + shift[o])
// A point-major [MTOT][512] bf16, W row-major [512][512] bf16.
// Block tile 128x128, 8 waves (4 along M x 2 along N), wave = 2x4 C-tiles.
// Stage layout (byte offsets in dynamic LDS):
//   stage s: A tile at s*16384 (8KB, [128][32] bf16), W tile at s*16384+8192.
// F32T=false: store bf16 point-major (feeds next layer).
// F32T=true : transpose via padded LDS and store f32 in (B,512,NP) layout.
// ---------------------------------------------------------------------------
template <bool F32T>
__global__ __launch_bounds__(256) void gemm_bn_relu_kernel(
        const bf16* __restrict__ A, const bf16* __restrict__ W,
        const float* __restrict__ scale, const float* __restrict__ shift,
        bf16* __restrict__ outB, float* __restrict__ outF) {
    extern __shared__ char smem[];
    float* lds_t = (float*)smem;           // [128][132] (layer-2 epilogue only)

    const int t    = threadIdx.x;
    const int lane = t & 31;
    const int w    = t >> 5;
    const int wm   = w & 3;        // wave position along M (points)
    const int wn   = w >> 2;       // wave position along N (out channels)
    const int l16  = lane & 15;
    const int lh   = lane >> 4;
    const int mbase = blockIdx.x * 128;
    const int nbase = blockIdx.y * 128;

    // per-thread copy slots: two 16B chunks of each tile
    // chunk cid (0..511): row = cid/4, 16B sub-chunk = cid%4
    const int r0c = t >> 2,         c0c = (t & 3) * 8;
    const int r1c = (t + 256) >> 2, c1c = ((t + 256) & 3) * 8;

    auto issue_stage = [&](int stage, int k0) {
        unsigned base = (unsigned)(uintptr_t)(smem) + stage * 16384u;
        async_copy_b128(base + (r0c * 32 + c0c) * 2,
                        A + (size_t)(mbase + r0c) * CIN + k0 + c0c);
        async_copy_b128(base + (r1c * 32 + c1c) * 2,
                        A + (size_t)(mbase + r1c) * CIN + k0 + c1c);
        async_copy_b128(base + 8192u + (r0c * 32 + c0c) * 2,
                        W + (size_t)(nbase + r0c) * CIN + k0 + c0c);
        async_copy_b128(base + 8192u + (r1c * 32 + c1c) * 2,
                        W + (size_t)(nbase + r1c) * CIN + k0 + c1c);
    };

    v8f acc[2][4] = {};
    issue_stage(0, 0);

    for (int k = 0; k < 16; ++k) {
        const int cur = k & 1;
        wait_async0();             // my stage-cur copies have landed in LDS
        __syncthreads();           // everyone's copies landed; prev reads done
        if (k < 15) issue_stage(cur ^ 1, (k + 1) * 32);

        const bf16* lds_a = (const bf16*)(smem + cur * 16384);
        const bf16* lds_b = lds_a + 128 * 32;

        Frag fa[2], fb[4];
        #pragma unroll
        for (int tm = 0; tm < 2; ++tm) {
            const bf16* pa = lds_a + (wm * 32 + tm * 16 + l16) * 32;
            fa[tm].h[0] = *(const v8bf*)(pa + lh * 8);        // K 0-7 / 8-15
            fa[tm].h[1] = *(const v8bf*)(pa + 16 + lh * 8);   // K 16-23 / 24-31
        }
        #pragma unroll
        for (int tn = 0; tn < 4; ++tn) {
            const bf16* pb = lds_b + (wn * 64 + tn * 16 + l16) * 32;
            fb[tn].h[0] = *(const v8bf*)(pb + lh * 8);        // K 0-7 / 8-15
            fb[tn].h[1] = *(const v8bf*)(pb + 16 + lh * 8);   // K 16-23 / 24-31
        }
        #pragma unroll
        for (int tm = 0; tm < 2; ++tm)
            #pragma unroll
            for (int tn = 0; tn < 4; ++tn)
                acc[tm][tn] = __builtin_amdgcn_wmma_f32_16x16x32_bf16(
                    false, fa[tm].v, false, fb[tn].v,
                    (short)0, acc[tm][tn], false, false);
        // no trailing barrier: next iteration's wait+barrier protects the
        // buffer being refilled (a wave passing that barrier has completed
        // its fragment ds_loads, which are waited before the WMMAs above).
    }

    if (!F32T) {
        // bf16 point-major store; lanes 0-15 cover 16 contiguous channels
        #pragma unroll
        for (int tn = 0; tn < 4; ++tn) {
            int o = nbase + wn * 64 + tn * 16 + l16;
            float sc = scale[o], sh = shift[o];
            #pragma unroll
            for (int tm = 0; tm < 2; ++tm) {
                int rowg = mbase + wm * 32 + tm * 16 + lh * 8;
                #pragma unroll
                for (int r = 0; r < 8; ++r) {
                    float y = fmaxf(acc[tm][tn][r] * sc + sh, 0.0f);
                    outB[(size_t)(rowg + r) * OUT + o] = (bf16)y;
                }
            }
        }
    } else {
        __syncthreads();  // all waves done with GEMM stages before LDS reuse
        // transpose 128x128 f32 block through padded LDS, then 512B row stores
        #pragma unroll
        for (int tn = 0; tn < 4; ++tn) {
            int o  = nbase + wn * 64 + tn * 16 + l16;
            int ol = wn * 64 + tn * 16 + l16;
            float sc = scale[o], sh = shift[o];
            #pragma unroll
            for (int tm = 0; tm < 2; ++tm) {
                int pl = wm * 32 + tm * 16 + lh * 8;
                #pragma unroll
                for (int r = 0; r < 8; ++r) {
                    float y = fmaxf(acc[tm][tn][r] * sc + sh, 0.0f);
                    lds_t[ol * 132 + pl + r] = y;
                }
            }
        }
        __syncthreads();
        const int bb = mbase >> 13;          // batch (8192 rows per batch)
        const int pb = mbase & (NP - 1);
        #pragma unroll
        for (int pass = 0; pass < 16; ++pass) {
            int ol = (t >> 5) + pass * 8;
            int p4 = (t & 31) * 4;
            v4f val = *(const v4f*)(lds_t + ol * 132 + p4);
            *(v4f*)(outF + ((size_t)bb * OUT + nbase + ol) * NP + pb + p4) = val;
        }
    }
}

// ---------------------------------------------------------------------------
extern "C" void kernel_launch(void* const* d_in, const int* in_sizes, int n_in,
                              void* d_out, int out_size, void* d_ws, size_t ws_size,
                              hipStream_t stream) {
    const float* unknown      = (const float*)d_in[0];
    const float* known        = (const float*)d_in[1];
    const float* unknow_feats = (const float*)d_in[2];
    const float* known_feats  = (const float*)d_in[3];
    const float* W1 = (const float*)d_in[4];
    const float* g1 = (const float*)d_in[5];
    const float* b1 = (const float*)d_in[6];
    const float* m1 = (const float*)d_in[7];
    const float* v1 = (const float*)d_in[8];
    const float* W2 = (const float*)d_in[9];
    const float* g2 = (const float*)d_in[10];
    const float* b2 = (const float*)d_in[11];
    const float* m2 = (const float*)d_in[12];
    const float* v2 = (const float*)d_in[13];
    float* out = (float*)d_out;

    // workspace carve-up (all 256B aligned)
    char* ws = (char*)d_ws;
    int*   idxb = (int*)   (ws + 0);          // 786432 B
    float* wgtb = (float*) (ws + 786432);     // 786432 B
    bf16*  W1b  = (bf16*)  (ws + 1572864);    // 524288 B
    bf16*  W2b  = (bf16*)  (ws + 2097152);    // 524288 B
    float* s1   = (float*) (ws + 2621440);
    float* sh1  = (float*) (ws + 2623488);
    float* s2   = (float*) (ws + 2625536);
    float* sh2  = (float*) (ws + 2627584);
    bf16*  XinT = (bf16*)  (ws + 2629632);    // 67108864 B
    bf16*  HT   = (bf16*)  (ws + 69738496);   // 67108864 B

    // 1) BN fold + weight conversion
    prep_kernel<<<dim3((2 * CIN * OUT + 2 * OUT + 255) / 256), dim3(256), 0, stream>>>(
        W1, W2, g1, b1, m1, v1, g2, b2, m2, v2, W1b, W2b, s1, sh1, s2, sh2);

    // 2) 3-NN + weights
    three_nn_kernel<<<dim3(NP / 256, BB), dim3(256), 0, stream>>>(
        unknown, known, idxb, wgtb);

    // 3) interpolate + concat -> bf16 point-major input
    build_input_kernel<<<dim3(NP / 16, BB), dim3(256), 0, stream>>>(
        known_feats, unknow_feats, idxb, wgtb, XinT);

    // 4) layer 1: WMMA GEMM + BN + ReLU -> bf16 point-major (32KB LDS stages)
    gemm_bn_relu_kernel<false><<<dim3(MTOT / 128, OUT / 128), dim3(256),
                                 2 * 16384, stream>>>(
        XinT, W1b, s1, sh1, HT, nullptr);

    // 5) layer 2: WMMA GEMM + BN + ReLU -> f32 (B,512,NP) via LDS transpose
    gemm_bn_relu_kernel<true><<<dim3(MTOT / 128, OUT / 128), dim3(256),
                                128 * 132 * 4, stream>>>(
        HT, W2b, s2, sh2, nullptr, out);
}
